// CGConvLayer_32066225832046
// MI455X (gfx1250) — matrix operand order repaired and checked by reading, not compiled, not run on hardware
//
#include <hip/hip_runtime.h>
#include <hip/hip_bf16.h>
#include <math.h>

typedef __attribute__((ext_vector_type(16))) __bf16 v16bf;
typedef __attribute__((ext_vector_type(8)))  __bf16 v8bf;
typedef __attribute__((ext_vector_type(4)))  __bf16 v4bf;
typedef __attribute__((ext_vector_type(8)))  float  v8f;

#define NODE_DIM 96
#define EDGE_DIM 64
#define IN_DIM   256
#define ROW_PITCH 264          // halfs: 256 + 8 pad -> row start shifts 16B = 4 banks per row
#define NT_TILES 6             // 96 / 16
#define KB_STEPS 8             // 256 / 32
#define FRAG_HALFS (KB_STEPS * NT_TILES * 32 * 16)  // 24576 bf16 per weight matrix

__device__ __forceinline__ void cvt_store4(__bf16* p, float4 v) {
  v4bf b;
  b[0] = (__bf16)v.x; b[1] = (__bf16)v.y; b[2] = (__bf16)v.z; b[3] = (__bf16)v.w;
  *(v4bf*)p = b;   // ds_store_b64
}

// ---- repack W[256x96] f32 -> bf16 B-fragments matching CDNA5 16-bit B 32x16 layout ----
// frag index: (((nt*8 + kb)*32 + lane)*16 + j2); K = kb*32 + 8*(lane>>4) + (j2&7) + 16*(j2>>3)
__global__ void __launch_bounds__(256) prep_kernel(const float* __restrict__ We,
                                                   const float* __restrict__ Wn,
                                                   __bf16* __restrict__ fe,
                                                   __bf16* __restrict__ fn) {
  int t = blockIdx.x * 256 + threadIdx.x;
  if (t >= 2 * FRAG_HALFS) return;
  int w   = t >= FRAG_HALFS;
  int idx = w ? t - FRAG_HALFS : t;
  int j2   = idx & 15;
  int lane = (idx >> 4) & 31;
  int kb   = (idx >> 9) & 7;
  int nt   = idx >> 12;
  int n = nt * 16 + (lane & 15);
  int k = kb * 32 + ((lane >> 4) << 3) + (j2 & 7) + ((j2 >> 3) << 4);
  float val = (w ? Wn : We)[k * NODE_DIM + n];
  (w ? fn : fe)[idx] = (__bf16)val;
}

__global__ void __launch_bounds__(256) zero_kernel(float* __restrict__ p, size_t total) {
  size_t t = (size_t)blockIdx.x * 256 + threadIdx.x;
  if (t < total) p[t] = 0.0f;
}

__global__ void __launch_bounds__(256) count_kernel(const int* __restrict__ dst,
                                                    float* __restrict__ cnt, int E) {
  int e = blockIdx.x * 256 + threadIdx.x;
  if (e < E) atomicAdd(&cnt[dst[e]], 1.0f);
}

// ---- main WMMA kernel: 4 waves/block, one 16-edge tile per wave ----
__global__ void __launch_bounds__(128) edge_gemm_kernel(
    const float* __restrict__ h, const int* __restrict__ ei, const float* __restrict__ ea,
    const __bf16* __restrict__ fe, const __bf16* __restrict__ fn,
    const float* __restrict__ be, const float* __restrict__ bn,
    float* __restrict__ acc, int E)
{
  __shared__ __bf16 lds[4 * 16 * ROW_PITCH];
  const int tid  = threadIdx.x;
  const int wave = tid >> 5;
  const int lane = tid & 31;
  const int tiles = (E + 15) >> 4;
  const int tileIdx = blockIdx.x * 4 + wave;
  const int te = tileIdx << 4;
  const int* __restrict__ srcIdx = ei;
  const int* __restrict__ dstIdx = ei + E;
  __bf16* tile = lds + wave * 16 * ROW_PITCH;
  const bool active = tileIdx < tiles;

  // ---- stage A tile: concat(h[src], h[dst], edge_attr) as bf16; 2 lanes per row ----
  if (active) {
    const int r = lane >> 1;
    const int half = lane & 1;
    int e = te + r; if (e >= E) e = E - 1;
    const int s = srcIdx[e];
    const int d = dstIdx[e];
    __bf16* row = tile + r * ROW_PITCH;
    const float4* hs = (const float4*)(h + (size_t)s * NODE_DIM);
    const float4* hd = (const float4*)(h + (size_t)d * NODE_DIM);
    if (half == 0) {                       // cols 0..127
      #pragma unroll
      for (int i = 0; i < 24; ++i) cvt_store4(row + i * 4, hs[i]);          // h[src][0..95]
      #pragma unroll
      for (int i = 0; i < 8;  ++i) cvt_store4(row + 96 + i * 4, hd[i]);     // h[dst][0..31]
    } else {                               // cols 128..255
      #pragma unroll
      for (int i = 0; i < 16; ++i) cvt_store4(row + 128 + i * 4, hd[8 + i]); // h[dst][32..95]
      const float4* ev = (const float4*)(ea + (size_t)e * EDGE_DIM);
      #pragma unroll
      for (int i = 0; i < 16; ++i) cvt_store4(row + 192 + i * 4, ev[i]);     // edge_attr[0..63]
    }
  }
  __syncthreads();
  if (!active) return;

  // per-row destination nodes (C layout: lane l, vgpr v -> row m = v + 8*(l>>4))
  int dRow[8];
  #pragma unroll
  for (int v = 0; v < 8; ++v) {
    int e = te + v + ((lane >> 4) << 3);
    dRow[v] = (e < E) ? dstIdx[e] : -1;
  }

  // A-fragment base: lane<16 reads K = kb*32 + [0..7] & [16..23]; lane>=16: +8
  const __bf16* aBase = tile + (lane & 15) * ROW_PITCH + ((lane >> 4) << 3);

  for (int nt = 0; nt < NT_TILES; ++nt) {
    v8f cg = {};
    v8f cm = {};
    #pragma unroll
    for (int kb = 0; kb < KB_STEPS; ++kb) {
      const __bf16* ap = aBase + kb * 32;
      v8bf a0 = *(const v8bf*)ap;          // ds_load_b128
      v8bf a1 = *(const v8bf*)(ap + 16);   // ds_load_b128
      union { v16bf v; v8bf h2[2]; } au;
      au.h2[0] = a0; au.h2[1] = a1;
      const size_t fo = (((size_t)nt * KB_STEPS + kb) * 32 + lane) * 16;
      v16bf bE = *(const v16bf*)(fe + fo); // 2x global_load_b128, L2/WGP$ resident
      v16bf bN = *(const v16bf*)(fn + fo);
      cg = __builtin_amdgcn_wmma_f32_16x16x32_bf16(false, au.v, false, bE,
                                                   (short)0, cg, false, false);
      cm = __builtin_amdgcn_wmma_f32_16x16x32_bf16(false, au.v, false, bN,
                                                   (short)0, cm, false, false);
    }
    const int n = nt * 16 + (lane & 15);
    const float bge = be[n];
    const float bgn = bn[n];
    #pragma unroll
    for (int v = 0; v < 8; ++v) {
      float g  = 1.0f / (1.0f + __expf(-(cg[v] + bge)));                  // sigmoid
      float x  = cm[v] + bgn;
      float sp = fmaxf(x, 0.0f) + log1pf(__expf(-fabsf(x)));              // stable softplus
      if (dRow[v] >= 0)
        atomicAdd(&acc[(size_t)dRow[v] * NODE_DIM + n], g * sp);          // segment sum
    }
  }
}

__global__ void __launch_bounds__(256) finalize_kernel(
    const float* __restrict__ h, const float* __restrict__ acc,
    const float* __restrict__ cnt, float* __restrict__ out, size_t total) {
  size_t t = (size_t)blockIdx.x * 256 + threadIdx.x;
  if (t < total) {
    size_t node = t / NODE_DIM;
    float c = fmaxf(cnt[node], 1.0f);
    out[t] = h[t] + acc[t] / c;
  }
}

extern "C" void kernel_launch(void* const* d_in, const int* in_sizes, int n_in,
                              void* d_out, int out_size, void* d_ws, size_t ws_size,
                              hipStream_t stream) {
  const float* h  = (const float*)d_in[0];
  const int*   ei = (const int*)d_in[1];   // [2, E] (src row, then dst row)
  const float* ea = (const float*)d_in[2];
  const float* We = (const float*)d_in[3];
  const float* be = (const float*)d_in[4];
  const float* Wn = (const float*)d_in[5];
  const float* bn = (const float*)d_in[6];
  float* out = (float*)d_out;

  const int N = in_sizes[0] / NODE_DIM;
  const int E = in_sizes[1] / 2;

  // workspace layout
  __bf16* fe = (__bf16*)d_ws;
  __bf16* fn = fe + FRAG_HALFS;
  float*  acc = (float*)((char*)d_ws + (size_t)2 * FRAG_HALFS * sizeof(__bf16));
  float*  cnt = acc + (size_t)N * NODE_DIM;

  // 1) pack weights into WMMA B-fragments (96 KB total, L2-resident)
  {
    int total = 2 * FRAG_HALFS;
    prep_kernel<<<(total + 255) / 256, 256, 0, stream>>>(We, Wn, fe, fn);
  }
  // 2) zero accumulator + counts (contiguous N*96 + N floats)
  {
    size_t total = (size_t)N * (NODE_DIM + 1);
    zero_kernel<<<(int)((total + 255) / 256), 256, 0, stream>>>(acc, total);
  }
  // 3) in-degree counts
  count_kernel<<<(E + 255) / 256, 256, 0, stream>>>(ei + E, cnt, E);
  // 4) WMMA edge GEMM + gated epilogue + atomic segment-sum
  {
    int tiles  = (E + 15) / 16;
    int blocks = (tiles + 3) / 4;
    edge_gemm_kernel<<<blocks, 128, 0, stream>>>(h, ei, ea, fe, fn, be, bn, acc, E);
  }
  // 5) residual + degree normalization
  {
    size_t total = (size_t)N * NODE_DIM;
    finalize_kernel<<<(int)((total + 255) / 256), 256, 0, stream>>>(h, acc, cnt, out, total);
  }
}
